// ReckoningLM_23124103922107
// MI455X (gfx1250) — matrix operand order repaired and checked by reading, not compiled, not run on hardware
//
#include <hip/hip_runtime.h>
#include <hip/hip_bf16.h>
#include <math.h>

// ---------------------------------------------------------------------------
// Problem constants (match reference)
// ---------------------------------------------------------------------------
constexpr int Bsz  = 8;
constexpr int Tsz  = 256;
constexpr int Mtok = Bsz * Tsz;     // 2048 token rows
constexpr int Dd   = 1024;
constexpr int Lx   = 8;
constexpr int NCc  = 4096;
constexpr int DCc  = 256;
constexpr int Kk   = 8;
constexpr int DFFf = 4096;
constexpr int VOC  = 50257;

typedef __bf16 bf16_t;
typedef __bf16 v16bf __attribute__((ext_vector_type(16)));
typedef float  v8f   __attribute__((ext_vector_type(8)));

union FragAB { uint4 u[2]; v16bf v; };

// ---------------------------------------------------------------------------
// CDNA5 async copy: global -> LDS, 16 bytes per lane, tracked by ASYNCcnt.
// LDS aperture maps flat addr[31:0] to the LDS byte offset (ISA 10.2), so the
// truncated flat pointer is a valid VDST LDS address.
// ---------------------------------------------------------------------------
__device__ __forceinline__ void async_ld16(void* lds, const void* g) {
  uint32_t l = (uint32_t)(uintptr_t)lds;
  uint64_t a = (uint64_t)(uintptr_t)g;
  asm volatile("global_load_async_to_lds_b128 %0, %1, off"
               :: "v"(l), "v"(a) : "memory");
}
__device__ __forceinline__ void wait_async0() {
  asm volatile("s_wait_asynccnt 0x0" ::: "memory");
}

// ---------------------------------------------------------------------------
// Helpers
// ---------------------------------------------------------------------------
__device__ __forceinline__ float block_reduce_sum(float v, float* red) {
  const int tid = threadIdx.x;
  red[tid] = v;
  __syncthreads();
  #pragma unroll
  for (int s = 128; s > 0; s >>= 1) {
    if (tid < s) red[tid] += red[tid + s];
    __syncthreads();
  }
  float r = red[0];
  __syncthreads();   // make `red` reusable
  return r;
}

__device__ __forceinline__ void st4bf16(bf16_t* p, float4 v) {
  union { bf16_t h[4]; uint2 u; } t;
  t.h[0] = (bf16_t)v.x; t.h[1] = (bf16_t)v.y;
  t.h[2] = (bf16_t)v.z; t.h[3] = (bf16_t)v.w;
  *reinterpret_cast<uint2*>(p) = t.u;
}

__device__ __forceinline__ float sigmoidf_(float z) {
  return 1.f / (1.f + expf(-z));
}

// ---------------------------------------------------------------------------
// Generic bf16 WMMA GEMM:  C[M,N] = alpha * A_bf16[M,K] x B_f32 (+ C if BETA)
//   BNK == true  : B stored as [N][K] (k-contiguous rows: W_route, tied head)
//   BNK == false : B stored as [K][N] (n-contiguous rows: W_in/out/g/u/d)
// Block tile 128x128x32, 256 threads = 8 waves (2 x 4), wave tile 64x32.
// Double-buffered LDS; A staged with global_load_async_to_lds_b128, B staged
// through VGPRs (fp32 -> bf16 conversion) overlapping the WMMA compute.
// Fully specialized over <BNK, BETA> so the k-loop has no runtime branches.
// ---------------------------------------------------------------------------
#define TILE_M 128
#define TILE_N 128
#define TILE_K 32

__device__ __forceinline__ void a_issue_async(const bf16_t* A, int Kdim,
                                              int m0, int k0,
                                              bf16_t* Asbuf, int tid) {
  #pragma unroll
  for (int i = 0; i < 2; ++i) {
    int q   = tid + i * 256;           // 0..511
    int row = q >> 2;                  // 0..127
    int oct = q & 3;                   // 8 bf16 per oct
    async_ld16(&Asbuf[row * TILE_K + oct * 8],
               &A[(size_t)(m0 + row) * Kdim + k0 + oct * 8]);
  }
}

template <bool BNK>
__device__ __forceinline__ void b_issue(const float* B, int Ndim, int Kdim,
                                        int n0, int k0, int tid, float4 r[4]) {
  if (BNK) {
    #pragma unroll
    for (int i = 0; i < 4; ++i) {
      int q  = tid + i * 256;          // 0..1023
      int n  = q >> 3;                 // 0..127
      int kq = q & 7;                  // 4 floats per kq
      r[i] = make_float4(0.f, 0.f, 0.f, 0.f);
      if (n0 + n < Ndim)
        r[i] = *reinterpret_cast<const float4*>(
            &B[(size_t)(n0 + n) * Kdim + k0 + kq * 4]);
    }
  } else {
    #pragma unroll
    for (int i = 0; i < 4; ++i) {
      int q  = tid + i * 256;
      int kk = q >> 5;                 // 0..31
      int n  = (q & 31) * 4;
      r[i] = *reinterpret_cast<const float4*>(
          &B[(size_t)(k0 + kk) * Ndim + n0 + n]);
    }
  }
}

template <bool BNK>
__device__ __forceinline__ void b_commit(bf16_t* Bsbuf, int tid,
                                         const float4 r[4]) {
  if (BNK) {
    #pragma unroll
    for (int i = 0; i < 4; ++i) {
      int q  = tid + i * 256;
      int n  = q >> 3;
      int kq = q & 7;
      st4bf16(&Bsbuf[n * TILE_K + kq * 4], r[i]);
    }
  } else {
    #pragma unroll
    for (int i = 0; i < 4; ++i) {
      int q  = tid + i * 256;
      int kk = q >> 5;
      int n  = (q & 31) * 4;
      Bsbuf[(n + 0) * TILE_K + kk] = (bf16_t)r[i].x;
      Bsbuf[(n + 1) * TILE_K + kk] = (bf16_t)r[i].y;
      Bsbuf[(n + 2) * TILE_K + kk] = (bf16_t)r[i].z;
      Bsbuf[(n + 3) * TILE_K + kk] = (bf16_t)r[i].w;
    }
  }
}

template <bool BNK, bool BETA>
__global__ __launch_bounds__(256) void gemm_bf16_wmma(
    const bf16_t* __restrict__ A, const float* __restrict__ B,
    float* __restrict__ C, int Ndim, int Kdim, float alpha)
{
  __shared__ bf16_t As[2][TILE_M * TILE_K];   // [m][k], k contiguous
  __shared__ bf16_t Bs[2][TILE_N * TILE_K];   // [n][k], k contiguous

  const int tid  = threadIdx.x;
  const int lane = tid & 31;
  const int wid  = tid >> 5;
  const int wm   = wid >> 2;      // 0..1  -> 64 rows each
  const int wn   = wid & 3;       // 0..3  -> 32 cols each
  const int l16  = lane & 15;
  const int half = lane >> 4;     // lanes 16..31 are the "upper half"

  const int m0 = blockIdx.y * TILE_M;
  const int n0 = blockIdx.x * TILE_N;

  v8f acc[4][2];
  #pragma unroll
  for (int mi = 0; mi < 4; ++mi)
    #pragma unroll
    for (int ni = 0; ni < 2; ++ni)
      #pragma unroll
      for (int r = 0; r < 8; ++r)
        acc[mi][ni][r] = 0.0f;

  const int ktiles = Kdim / TILE_K;

  // ---- prologue: stage tile 0 ---------------------------------------------
  {
    float4 r[4];
    a_issue_async(A, Kdim, m0, 0, As[0], tid);
    b_issue<BNK>(B, Ndim, Kdim, n0, 0, tid, r);
    b_commit<BNK>(Bs[0], tid, r);
    wait_async0();
    __syncthreads();
  }

  for (int kt = 0; kt < ktiles; ++kt) {
    const int cur = kt & 1;
    const int nxt = cur ^ 1;
    const bool has_next = (kt + 1 < ktiles);

    float4 rb[4];
    if (has_next) {
      const int k1 = (kt + 1) * TILE_K;
      a_issue_async(A, Kdim, m0, k1, As[nxt], tid);
      b_issue<BNK>(B, Ndim, Kdim, n0, k1, tid, rb);
      if (kt + 2 < ktiles) {
        // prefetch the weight stream two tiles ahead (global_prefetch)
        const int k2 = (kt + 2) * TILE_K;
        if (BNK)
          __builtin_prefetch(&B[(size_t)(n0 + (tid >> 3)) * Kdim + k2], 0, 2);
        else
          __builtin_prefetch(&B[(size_t)(k2 + (tid >> 5)) * Ndim + n0], 0, 2);
      }
    }

    // ---- fragments (ISA 7.12.2 16-bit A 16x32 / B 32x16 layouts) ----------
    FragAB afr[4], bfr[2];
    #pragma unroll
    for (int mi = 0; mi < 4; ++mi) {
      const uint4* pa = reinterpret_cast<const uint4*>(
          &As[cur][(wm * 64 + mi * 16 + l16) * TILE_K]);
      // VGPR0-3: K = half*8 .. +7 ; VGPR4-7: K = 16 + half*8 .. +7
      afr[mi].u[0] = pa[half];
      afr[mi].u[1] = pa[2 + half];
    }
    #pragma unroll
    for (int ni = 0; ni < 2; ++ni) {
      const uint4* pb = reinterpret_cast<const uint4*>(
          &Bs[cur][(wn * 32 + ni * 16 + l16) * TILE_K]);
      // VGPR0-3: K = half*8 .. +7 ; VGPR4-7: K = 16 + half*8 .. +7
      bfr[ni].u[0] = pb[half];
      bfr[ni].u[1] = pb[2 + half];
    }
    #pragma unroll
    for (int mi = 0; mi < 4; ++mi)
      #pragma unroll
      for (int ni = 0; ni < 2; ++ni)
        acc[mi][ni] = __builtin_amdgcn_wmma_f32_16x16x32_bf16(
            false, afr[mi].v, false, bfr[ni].v, (short)0, acc[mi][ni],
            false, false);

    if (has_next) {
      b_commit<BNK>(Bs[nxt], tid, rb);
      wait_async0();
    }
    __syncthreads();
  }

  // ---- epilogue: C/D layout (VGPR r -> M = half*8 + r, N = lane%16) -------
  #pragma unroll
  for (int mi = 0; mi < 4; ++mi) {
    #pragma unroll
    for (int ni = 0; ni < 2; ++ni) {
      const int col = n0 + wn * 32 + ni * 16 + l16;
      if (col < Ndim) {
        const int rowb = m0 + wm * 64 + mi * 16 + half * 8;
        #pragma unroll
        for (int r = 0; r < 8; ++r) {
          size_t idx = (size_t)(rowb + r) * Ndim + col;
          float v = acc[mi][ni][r] * alpha;
          if (BETA) v += C[idx];
          C[idx] = v;
        }
      }
    }
  }
}

// ---------------------------------------------------------------------------
// Embedding gather + RMSNorm (one block per token row)
// ---------------------------------------------------------------------------
__global__ __launch_bounds__(256) void embed_rmsnorm(
    const int* __restrict__ tokens, const float* __restrict__ embed,
    const float* __restrict__ w, float* __restrict__ x)
{
  __shared__ float red[256];
  const int row = blockIdx.x;
  const int tok = tokens[row];
  const float* e = embed + (size_t)tok * Dd;
  float v[4]; float p = 0.f;
  #pragma unroll
  for (int j = 0; j < 4; ++j) {
    v[j] = e[threadIdx.x + j * 256];
    p += v[j] * v[j];
  }
  float r = rsqrtf(block_reduce_sum(p, red) / (float)Dd + 1e-6f);
  #pragma unroll
  for (int j = 0; j < 4; ++j) {
    int d = threadIdx.x + j * 256;
    x[(size_t)row * Dd + d] = v[j] * r * w[d];
  }
}

// ---------------------------------------------------------------------------
// RMSNorm(fp32) -> bf16 (final head input)
// ---------------------------------------------------------------------------
__global__ __launch_bounds__(256) void rms_to_bf16(
    const float* __restrict__ x, const float* __restrict__ w,
    bf16_t* __restrict__ o)
{
  __shared__ float red[256];
  const int row = blockIdx.x;
  const size_t base = (size_t)row * Dd;
  float v[4]; float p = 0.f;
  #pragma unroll
  for (int j = 0; j < 4; ++j) {
    v[j] = x[base + threadIdx.x + j * 256];
    p += v[j] * v[j];
  }
  float r = rsqrtf(block_reduce_sum(p, red) / (float)Dd + 1e-6f);
  #pragma unroll
  for (int j = 0; j < 4; ++j) {
    int d = threadIdx.x + j * 256;
    o[base + d] = (bf16_t)(v[j] * r * w[d]);
  }
}

// ---------------------------------------------------------------------------
// Top-8 routing + softmax + weighted cell readout. One wave32 per token.
// ---------------------------------------------------------------------------
__global__ void topk_readout(const float* __restrict__ scores,
                             const float* __restrict__ cells,
                             float* __restrict__ rout)
{
  const int t    = blockIdx.x;
  const int lane = threadIdx.x;              // 0..31
  const float* s = scores + (size_t)t * NCc;

  int   cidx[Kk];
  float cval[Kk];
  for (int p = 0; p < Kk; ++p) {
    float best = -3.4e38f; int bidx = NCc;
    for (int i = lane; i < NCc; i += 32) {
      bool skip = false;
      #pragma unroll
      for (int j = 0; j < Kk; ++j)
        if (j < p) skip |= (cidx[j] == i);
      float v = s[i];
      if (!skip && (v > best || (v == best && i < bidx))) { best = v; bidx = i; }
    }
    #pragma unroll
    for (int off = 16; off >= 1; off >>= 1) {
      float ov = __shfl_xor(best, off, 32);
      int   oi = __shfl_xor(bidx, off, 32);
      if (ov > best || (ov == best && oi < bidx)) { best = ov; bidx = oi; }
    }
    cidx[p] = bidx; cval[p] = best;
  }
  // softmax over the 8 selected scores (cval[0] is the max)
  float wk[Kk]; float wsum = 0.f;
  #pragma unroll
  for (int p = 0; p < Kk; ++p) { wk[p] = expf(cval[p] - cval[0]); wsum += wk[p]; }
  const float inv = 1.f / wsum;
  for (int dc = lane; dc < DCc; dc += 32) {
    float a = 0.f;
    #pragma unroll
    for (int p = 0; p < Kk; ++p)
      a += wk[p] * cells[(size_t)cidx[p] * DCc + dc];
    rout[(size_t)t * DCc + dc] = a * inv;
  }
}

// ---------------------------------------------------------------------------
// Gated decay state scan: one thread per (batch, d) — 8192 independent scans
// ---------------------------------------------------------------------------
__global__ __launch_bounds__(256) void state_scan(
    const float* __restrict__ x, float* __restrict__ state,
    const float* __restrict__ sd, const float* __restrict__ gw,
    const float* __restrict__ gb, const float* __restrict__ sc)
{
  const int idx = blockIdx.x * 256 + threadIdx.x;   // 0..8191
  const int b = idx >> 10;
  const int d = idx & (Dd - 1);
  float dec = sigmoidf_(sd[0]);
  dec = fminf(fmaxf(dec, 0.01f), 0.99f);
  const float w = gw[d], bb = gb[d], ss = sc[d];
  float h = 0.f;
  size_t base = (size_t)b * Tsz * Dd + d;
  for (int t = 0; t < Tsz; ++t) {
    float xv = x[base];
    h = h * dec + sigmoidf_(xv * w + bb) * xv;
    state[base] = h * ss;
    base += Dd;
  }
}

// ---------------------------------------------------------------------------
// Fused residual + double RMSNorm:
//   x = rmsnorm(x + cs*cell_out + state, nrm_w)   (fp32, in place)
//   h = bf16(rmsnorm(x, ffn_w))                   (FFN input)
// ---------------------------------------------------------------------------
__global__ __launch_bounds__(256) void combine_norm(
    float* __restrict__ x, const float* __restrict__ cell_out,
    const float* __restrict__ state, const float* __restrict__ cs,
    const float* __restrict__ nw, const float* __restrict__ fw,
    bf16_t* __restrict__ hb)
{
  __shared__ float red[256];
  const int row = blockIdx.x;
  const int tid = threadIdx.x;
  const float c = cs[0];
  const size_t base = (size_t)row * Dd;
  float y[4]; float p = 0.f;
  #pragma unroll
  for (int j = 0; j < 4; ++j) {
    int d = tid + j * 256;
    y[j] = x[base + d] + c * cell_out[base + d] + state[base + d];
    p += y[j] * y[j];
  }
  float r1 = rsqrtf(block_reduce_sum(p, red) / (float)Dd + 1e-6f);
  float p2 = 0.f;
  #pragma unroll
  for (int j = 0; j < 4; ++j) {
    int d = tid + j * 256;
    float xn = y[j] * r1 * nw[d];
    x[base + d] = xn;
    y[j] = xn;
    p2 += xn * xn;
  }
  float r2 = rsqrtf(block_reduce_sum(p2, red) / (float)Dd + 1e-6f);
  #pragma unroll
  for (int j = 0; j < 4; ++j) {
    int d = tid + j * 256;
    hb[base + d] = (bf16_t)(y[j] * r2 * fw[d]);
  }
}

// ---------------------------------------------------------------------------
// Elementwise kernels
// ---------------------------------------------------------------------------
__global__ void to_bf16(const float* __restrict__ s, bf16_t* __restrict__ d, int n) {
  int i = blockIdx.x * 256 + threadIdx.x;
  if (i < n) d[i] = (bf16_t)s[i];
}

__global__ void swiglu_bf16(const float* __restrict__ g, const float* __restrict__ u,
                            bf16_t* __restrict__ o, int n) {
  int i = blockIdx.x * 256 + threadIdx.x;
  if (i < n) {
    float gv = g[i];
    o[i] = (bf16_t)((gv / (1.f + expf(-gv))) * u[i]);
  }
}

__global__ void add_inplace(float* __restrict__ x, const float* __restrict__ a, int n) {
  int i = blockIdx.x * 256 + threadIdx.x;
  if (i < n) x[i] += a[i];
}

// ---------------------------------------------------------------------------
// Workspace layout (bytes) — ~121 MB total
// ---------------------------------------------------------------------------
constexpr size_t OFF_X     = 0;
constexpr size_t OFF_XB    = OFF_X     + (size_t)Mtok * Dd   * 4;
constexpr size_t OFF_STATE = OFF_XB    + (size_t)Mtok * Dd   * 2;
constexpr size_t OFF_SC    = OFF_STATE + (size_t)Mtok * Dd   * 4;  // scores / ffn gate
constexpr size_t OFF_U     = OFF_SC    + (size_t)Mtok * NCc  * 4;  // ffn up
constexpr size_t OFF_RIN   = OFF_U     + (size_t)Mtok * DFFf * 4;  // readout (+x_in)
constexpr size_t OFF_RINB  = OFF_RIN   + (size_t)Mtok * DCc  * 4;
constexpr size_t OFF_CO    = OFF_RINB  + (size_t)Mtok * DCc  * 2;  // cell_out / ff_out
constexpr size_t OFF_HB    = OFF_CO    + (size_t)Mtok * Dd   * 4;  // bf16 h (FFN / head)
constexpr size_t OFF_SWB   = OFF_HB    + (size_t)Mtok * Dd   * 2;  // bf16 swiglu

extern "C" void kernel_launch(void* const* d_in, const int* in_sizes, int n_in,
                              void* d_out, int out_size, void* d_ws, size_t ws_size,
                              hipStream_t stream) {
  (void)in_sizes; (void)n_in; (void)out_size; (void)ws_size;

  const int*   tokens  = (const int*)  d_in[0];
  const float* embed   = (const float*)d_in[1];
  const float* ln_in   = (const float*)d_in[2];
  const float* ln_out  = (const float*)d_in[3];
  const float* W_route = (const float*)d_in[4];
  const float* cells   = (const float*)d_in[5];
  const float* W_in    = (const float*)d_in[6];
  const float* W_out   = (const float*)d_in[7];
  const float* sdec    = (const float*)d_in[8];
  const float* sgw     = (const float*)d_in[9];
  const float* sgb     = (const float*)d_in[10];
  const float* sscale  = (const float*)d_in[11];
  const float* cscale  = (const float*)d_in[12];
  const float* W_g     = (const float*)d_in[13];
  const float* W_u     = (const float*)d_in[14];
  const float* W_d     = (const float*)d_in[15];
  const float* ffn_w   = (const float*)d_in[16];
  const float* nrm_w   = (const float*)d_in[17];

  char* ws = (char*)d_ws;
  float*  X    = (float*) (ws + OFF_X);
  bf16_t* XB   = (bf16_t*)(ws + OFF_XB);
  float*  ST   = (float*) (ws + OFF_STATE);
  float*  SC   = (float*) (ws + OFF_SC);
  float*  Uu   = (float*) (ws + OFF_U);
  float*  RIN  = (float*) (ws + OFF_RIN);
  bf16_t* RINB = (bf16_t*)(ws + OFF_RINB);
  float*  CO   = (float*) (ws + OFF_CO);
  bf16_t* HB   = (bf16_t*)(ws + OFF_HB);
  bf16_t* SWB  = (bf16_t*)(ws + OFF_SWB);

  const dim3 blk(256);
  const int rowsM = Mtok / 128;   // 16

  embed_rmsnorm<<<Mtok, blk, 0, stream>>>(tokens, embed, ln_in, X);

  for (int l = 0; l < Lx; ++l) {
    to_bf16<<<(Mtok * Dd) / 256, blk, 0, stream>>>(X, XB, Mtok * Dd);
    state_scan<<<(Bsz * Dd) / 256, blk, 0, stream>>>(
        X, ST, sdec + l, sgw + (size_t)l * Dd, sgb + (size_t)l * Dd,
        sscale + (size_t)l * Dd);

    // scores = (x @ Wr^T) / sqrt(D)   (Wr is [NC, D] -> B[N][K])
    gemm_bf16_wmma<true, false><<<dim3(NCc / 128, rowsM), blk, 0, stream>>>(
        XB, W_route + (size_t)l * NCc * Dd, SC, NCc, Dd, 0.03125f);
    topk_readout<<<Mtok, dim3(32), 0, stream>>>(
        SC, cells + (size_t)l * NCc * DCc, RIN);

    // RIN += x @ W_in    (accumulate onto readout)
    gemm_bf16_wmma<false, true><<<dim3(DCc / 128, rowsM), blk, 0, stream>>>(
        XB, W_in + (size_t)l * Dd * DCc, RIN, DCc, Dd, 1.f);
    to_bf16<<<(Mtok * DCc) / 256, blk, 0, stream>>>(RIN, RINB, Mtok * DCc);

    // cell_out = (readout + x_in) @ W_out
    gemm_bf16_wmma<false, false><<<dim3(Dd / 128, rowsM), blk, 0, stream>>>(
        RINB, W_out + (size_t)l * DCc * Dd, CO, Dd, DCc, 1.f);

    combine_norm<<<Mtok, blk, 0, stream>>>(
        X, CO, ST, cscale + l, nrm_w + (size_t)l * Dd,
        ffn_w + (size_t)l * Dd, HB);

    // SwiGLU FFN
    gemm_bf16_wmma<false, false><<<dim3(DFFf / 128, rowsM), blk, 0, stream>>>(
        HB, W_g + (size_t)l * Dd * DFFf, SC, DFFf, Dd, 1.f);
    gemm_bf16_wmma<false, false><<<dim3(DFFf / 128, rowsM), blk, 0, stream>>>(
        HB, W_u + (size_t)l * Dd * DFFf, Uu, DFFf, Dd, 1.f);
    swiglu_bf16<<<(Mtok * DFFf) / 256, blk, 0, stream>>>(SC, Uu, SWB, Mtok * DFFf);
    gemm_bf16_wmma<false, false><<<dim3(Dd / 128, rowsM), blk, 0, stream>>>(
        SWB, W_d + (size_t)l * DFFf * Dd, CO, Dd, DFFf, 1.f);
    add_inplace<<<(Mtok * Dd) / 256, blk, 0, stream>>>(X, CO, Mtok * Dd);
  }

  // logits = rmsnorm(h, ln_out) @ embed^T   (embed is [VOC, D] -> B[N][K])
  rms_to_bf16<<<Mtok, blk, 0, stream>>>(X, ln_out, HB);
  gemm_bf16_wmma<true, false><<<dim3((VOC + 127) / 128, rowsM), blk, 0, stream>>>(
      HB, embed, (float*)d_out, VOC, Dd, 1.f);
}
